// SnakeDecoderHead_28518582845736
// MI455X (gfx1250) — matrix operand order repaired and checked by reading, not compiled, not run on hardware
//
#include <hip/hip_runtime.h>
#include <cstdint>
#include <cstddef>

// ============================================================================
// SnakeDecoderHead on MI455X (gfx1250): wave32, WMMA bf16 (16x16x32, f32 acc).
//
//   [prep]  pack conv weights f32 -> bf16 im2col layout; fold BN params.
//   [k1]    bilinear sample -> init input rows (bf16, zero-padded).
//   [k2]    per-snake workgroup: head + 7 residual dilated circular convs as
//           WMMA GEMMs (M=256,N=128,K=9*Cin). Activations in LDS with 16-row
//           circular halo; dilation is a template param so B addressing is
//           linear; B offsets carried as ints (keeps ds_load_b128). B loads
//           are software-pipelined 2 tiles ahead so multiple ds_loads stay in
//           flight (partial s_wait_dscnt instead of full serialization).
//           Each wave accumulates BOTH its m-tiles per k-pass.
//   [k3]    fusion 1x1 GEMM + max-reduce, pred0/1 WMMA (B pipelined 3 ahead,
//           global), pred2 scalar, out = snakes + offset.
//
// d_in ordering assumption: top level insertion order (cnn_feature, snakes,
// params), params flattened alphabetically: fusion{b,w}, head{b,beta,gamma,
// mean,var,w}, pred0{b,w}, pred1{b,w}, pred2{b,w}, res0..6{b,beta,gamma,mean,
// var,w}.
// ============================================================================

typedef __attribute__((ext_vector_type(16))) __bf16 v16bf;
typedef __attribute__((ext_vector_type(8)))  float  v8f;

#define HROWS 160   // 16 halo + 128 main + 16 halo
#define HOFF  16    // main row p lives at LDS row (p + HOFF)

// ---------------- helpers ----------------

__device__ __forceinline__ uint16_t f2bf(float f) {
  uint32_t u = __float_as_uint(f);
  u += 0x7FFFu + ((u >> 16) & 1u);          // round-to-nearest-even
  return (uint16_t)(u >> 16);
}
__device__ __forceinline__ float bf2f(uint16_t h) {
  return __uint_as_float(((uint32_t)h) << 16);
}

__device__ __forceinline__ v16bf loadAB(const uint16_t* lo, const uint16_t* hi) {
  union { uint4 u[2]; v16bf v; } c;
  c.u[0] = *(const uint4*)lo;
  c.u[1] = *(const uint4*)hi;
  return c.v;
}

__device__ __forceinline__ v8f wmma_bf16(v16bf a, v16bf b, v8f c) {
  return __builtin_amdgcn_wmma_f32_16x16x32_bf16(
      /*neg_a=*/false, a, /*neg_b=*/false, b,
      /*c_mod=*/(short)0, c, /*reuse_a=*/false, /*reuse_b=*/false);
}

__device__ __forceinline__ v8f zero8() {
  v8f z;
#pragma unroll
  for (int r = 0; r < 8; ++r) z[r] = 0.0f;
  return z;
}

__device__ __forceinline__ uint4 pack8(const float* v) {
  uint4 r;
  r.x = (uint32_t)f2bf(v[0]) | ((uint32_t)f2bf(v[1]) << 16);
  r.y = (uint32_t)f2bf(v[2]) | ((uint32_t)f2bf(v[3]) << 16);
  r.z = (uint32_t)f2bf(v[4]) | ((uint32_t)f2bf(v[5]) << 16);
  r.w = (uint32_t)f2bf(v[6]) | ((uint32_t)f2bf(v[7]) << 16);
  return r;
}
__device__ __forceinline__ void unpack8(uint4 u, float* v) {
  v[0] = bf2f((uint16_t)(u.x & 0xffff)); v[1] = bf2f((uint16_t)(u.x >> 16));
  v[2] = bf2f((uint16_t)(u.y & 0xffff)); v[3] = bf2f((uint16_t)(u.y >> 16));
  v[4] = bf2f((uint16_t)(u.z & 0xffff)); v[5] = bf2f((uint16_t)(u.z >> 16));
  v[6] = bf2f((uint16_t)(u.w & 0xffff)); v[7] = bf2f((uint16_t)(u.w >> 16));
}

__device__ __forceinline__ void atomicMaxF(float* addr, float val) {
  int* ia = (int*)addr;
  int old = __float_as_int(*addr);
  while (__int_as_float(old) < val) {
    int assumed = old;
    old = atomicCAS(ia, assumed, __float_as_int(val));
    if (old == assumed) break;
  }
}

// ---------------- prep kernels ----------------

__global__ void pack_conv9(const float* __restrict__ src, uint16_t* __restrict__ dst,
                           int cin, int cinp) {
  int idx = blockIdx.x * 256 + threadIdx.x;
  int total = 256 * 9 * cinp;
  if (idx >= total) return;
  int o   = idx / (9 * cinp);
  int rem = idx - o * (9 * cinp);
  int tap = rem / cinp;
  int i   = rem - tap * cinp;
  float v = (i < cin) ? src[(o * cin + i) * 9 + tap] : 0.0f;
  dst[idx] = f2bf(v);
}

__global__ void pack_1x1(const float* __restrict__ src, uint16_t* __restrict__ dst,
                         int K, int srcStride, int total) {
  int idx = blockIdx.x * 256 + threadIdx.x;
  if (idx >= total) return;
  int m = idx / K;
  int k = idx - m * K;
  dst[idx] = f2bf(src[(size_t)m * srcStride + k]);
}

__global__ void pack_bn(const float* __restrict__ b, const float* __restrict__ gamma,
                        const float* __restrict__ beta, const float* __restrict__ mean,
                        const float* __restrict__ var, float* __restrict__ dst) {
  int c = threadIdx.x; // 256 threads
  float s = gamma[c] * rsqrtf(var[c] + 1e-5f);
  dst[c]       = b[c];
  dst[256 + c] = s;
  dst[512 + c] = beta[c] - mean[c] * s;
}

// ---------------- kernel 1: bilinear sampling ----------------
__global__ __launch_bounds__(256) void sample_kernel(
    const float* __restrict__ feat, const float* __restrict__ snakes,
    uint16_t* __restrict__ initT) {
  __shared__ float sx[128], sy[128];
  __shared__ float mn[2];
  const int t = blockIdx.x, tid = threadIdx.x;
  const float* sn = snakes + (size_t)t * 128 * 2;
  if (tid < 128) {
    sx[tid] = sn[tid * 2 + 0] * 0.25f;  // i_it = snakes / RO
    sy[tid] = sn[tid * 2 + 1] * 0.25f;
  }
  __syncthreads();
  if (tid == 0) {
    float ax = 3.4e38f, ay = 3.4e38f;
    for (int i = 0; i < 128; ++i) { ax = fminf(ax, sx[i]); ay = fminf(ay, sy[i]); }
    mn[0] = ax; mn[1] = ay;
  }
  __syncthreads();

  const int p = tid >> 1, half = tid & 1;
  uint16_t* dst = initT + ((size_t)t * 128 + p) * 256 + half * 128;
  if (half == 0) {
    float x = sx[p] - 0.5f, y = sy[p] - 0.5f;
    float x0f = floorf(x), y0f = floorf(y);
    float wx = x - x0f, wy = y - y0f;
    int x0 = (int)x0f, y0 = (int)y0f;
    int x1 = x0 + 1, y1 = y0 + 1;
    float vx0 = (x0 >= 0 && x0 < 256) ? 1.0f : 0.0f;
    float vx1 = (x1 >= 0 && x1 < 256) ? 1.0f : 0.0f;
    float vy0 = (y0 >= 0 && y0 < 256) ? 1.0f : 0.0f;
    float vy1 = (y1 >= 0 && y1 < 256) ? 1.0f : 0.0f;
    int x0c = min(max(x0, 0), 255), x1c = min(max(x1, 0), 255);
    int y0c = min(max(y0, 0), 255), y1c = min(max(y1, 0), 255);
    float w00 = (1.f - wx) * (1.f - wy) * vx0 * vy0;
    float w10 = wx * (1.f - wy) * vx1 * vy0;
    float w01 = (1.f - wx) * wy * vx0 * vy1;
    float w11 = wx * wy * vx1 * vy1;
    const float* fb = feat + (size_t)(t >> 4) * 128 * 65536;
    int i00 = y0c * 256 + x0c, i10 = y0c * 256 + x1c;
    int i01 = y1c * 256 + x0c, i11 = y1c * 256 + x1c;
    for (int c = 0; c < 128; ++c) {
      const float* fc = fb + (size_t)c * 65536;
      float v = w00 * fc[i00] + w10 * fc[i10] + w01 * fc[i01] + w11 * fc[i11];
      dst[c] = f2bf(v);
    }
  } else {
    dst[0] = f2bf(sn[p * 2 + 0] - 4.0f * mn[0]);  // c_it*RO = snakes - 4*min
    dst[1] = f2bf(sn[p * 2 + 1] - 4.0f * mn[1]);
    for (int c = 2; c < 128; ++c) dst[c] = 0;
  }
}

// ---------------- conv block GEMM, templated on <KC, D> ----------------
// M=256, N=128, K=9*KC. Haloed LDS, linear B addressing, int offsets (keeps
// ds_load), and B software-pipelined 2 tiles ahead.
template <int KC, int D>
__device__ __forceinline__ void conv_block(
    const uint16_t* __restrict__ W,     // [256][9*KC] bf16
    const uint16_t* xT,                 // LDS in  [HROWS][256] bf16 (haloed)
    uint16_t* yT,                       // LDS out [HROWS][256] bf16 (haloed)
    uint16_t* __restrict__ gout,        // stateT slice (+p*2048+m)
    const float* __restrict__ bias, const float* __restrict__ a1,
    const float* __restrict__ a0,
    bool residual, int wave, int lhalf, int l16)
{
  v8f acc0[8], acc1[8];
#pragma unroll
  for (int i = 0; i < 8; ++i) { acc0[i] = zero8(); acc1[i] = zero8(); }

  const int arow0 = (wave * 2) * 16 + l16;
  const uint16_t* wr0 = W + (size_t)arow0 * (9 * KC) + lhalf * 16;
  const uint16_t* wr1 = wr0 + (size_t)16 * (9 * KC);

  // B-row element offsets: start at tap 0 (row shift -4*D), advance D rows/tap
  int boff[8];
#pragma unroll
  for (int nt = 0; nt < 8; ++nt)
    boff[nt] = ((nt * 16 + l16) + HOFF - 4 * D) * 256 + lhalf * 16;

  for (int tap = 0; tap < 9; ++tap) {
    __builtin_prefetch(wr0 + KC, 0, 1);                // next-tap A streams
    __builtin_prefetch(wr1 + KC, 0, 1);
#pragma unroll
    for (int cb = 0; cb < KC; cb += 32) {
      v16bf A0 = loadAB(wr0 + cb, wr0 + cb + 8);
      v16bf A1 = loadAB(wr1 + cb, wr1 + cb + 8);
      // B pipeline: 2 tiles in flight
      v16bf B0 = loadAB(&xT[boff[0] + cb], &xT[boff[0] + cb + 8]);
      v16bf B1 = loadAB(&xT[boff[1] + cb], &xT[boff[1] + cb + 8]);
#pragma unroll
      for (int nt = 0; nt < 8; ++nt) {
        v16bf Bn = B1;
        if (nt < 6)
          Bn = loadAB(&xT[boff[nt + 2] + cb], &xT[boff[nt + 2] + cb + 8]);
        acc0[nt] = wmma_bf16(A0, B0, acc0[nt]);
        acc1[nt] = wmma_bf16(A1, B0, acc1[nt]);
        B0 = B1;
        B1 = Bn;
      }
    }
    wr0 += KC;
    wr1 += KC;
#pragma unroll
    for (int nt = 0; nt < 8; ++nt) boff[nt] += D * 256;   // next tap row
  }

  // epilogue: bias -> relu -> BN affine -> (+residual) -> LDS (+halo) + global
#pragma unroll
  for (int mi = 0; mi < 2; ++mi) {
    const v8f* acc = mi ? acc1 : acc0;
    const int mt = wave * 2 + mi;
    const int mrow = mt * 16 + lhalf * 8;
    float b8[8], s8[8], o8[8];
#pragma unroll
    for (int r = 0; r < 8; ++r) {
      b8[r] = bias[mrow + r]; s8[r] = a1[mrow + r]; o8[r] = a0[mrow + r];
    }
#pragma unroll
    for (int nt = 0; nt < 8; ++nt) {
      const int p = nt * 16 + l16;
      float v[8];
#pragma unroll
      for (int r = 0; r < 8; ++r) {
        float y = fmaxf(acc[nt][r] + b8[r], 0.0f);
        v[r] = y * s8[r] + o8[r];
      }
      if (residual) {
        float ov[8];
        unpack8(*(const uint4*)&xT[(p + HOFF) * 256 + mrow], ov);
#pragma unroll
        for (int r = 0; r < 8; ++r) v[r] += ov[r];
      }
      uint4 pk = pack8(v);
      *(uint4*)&yT[(p + HOFF) * 256 + mrow] = pk;
      if (nt == 0)   // p in [0,16): duplicate into high halo (row 144+p)
        *(uint4*)&yT[(p + 144) * 256 + mrow] = pk;
      if (nt == 7)   // p in [112,128): duplicate into low halo (row p-112)
        *(uint4*)&yT[(p - 112) * 256 + mrow] = pk;
      *(uint4*)(gout + (size_t)p * 2048 + mrow) = pk;
    }
  }
}

// ---------------- kernel 2: head + 7 residual circular convs ----------------
__global__ __launch_bounds__(256) void snake_net_blocks(
    const uint16_t* __restrict__ initT,   // [64][128][256]
    const uint16_t* __restrict__ WhPk,    // [256][9*160]
    const uint16_t* __restrict__ WrPk,    // [7][256][9*256]
    const float*    __restrict__ bnp,     // [8][3][256]
    uint16_t*       __restrict__ stateT)  // [64][128][2048]
{
  extern __shared__ uint16_t smem2[];
  uint16_t* xT = smem2;                   // [HROWS][256]
  uint16_t* yT = smem2 + HROWS * 256;
  const int t = blockIdx.x, tid = threadIdx.x;
  const int wave = tid >> 5, lane = tid & 31;
  const int lhalf = lane >> 4, l16 = lane & 15;

  { // load init input into xT (with circular halo duplication)
    const uint4* src = (const uint4*)(initT + (size_t)t * 128 * 256);
    uint4* dst = (uint4*)xT;
    for (int i = tid; i < HROWS * 32; i += 256) {   // 32 uint4 per row
      const int hrow = i >> 5;
      const int col  = i & 31;
      const int p = (hrow - HOFF) & 127;            // circular source row
      dst[i] = src[p * 32 + col];
    }
  }
  __syncthreads();

  for (int blk = 0; blk < 8; ++blk) {
    const float* bias = bnp + blk * 768;
    uint16_t* gout = stateT + (size_t)t * 128 * 2048 + blk * 256;
    if (blk == 0)
      conv_block<160, 1>(WhPk, xT, yT, gout, bias, bias + 256, bias + 512,
                         false, wave, lhalf, l16);
    else {
      const uint16_t* W = WrPk + (size_t)(blk - 1) * 256 * 2304;
      if (blk <= 3)
        conv_block<256, 1>(W, xT, yT, gout, bias, bias + 256, bias + 512,
                           true, wave, lhalf, l16);
      else if (blk <= 5)
        conv_block<256, 2>(W, xT, yT, gout, bias, bias + 256, bias + 512,
                           true, wave, lhalf, l16);
      else
        conv_block<256, 4>(W, xT, yT, gout, bias, bias + 256, bias + 512,
                           true, wave, lhalf, l16);
    }
    __syncthreads();
    uint16_t* tp = xT; xT = yT; yT = tp;   // swap double buffer
  }
}

// ---------------- tail GEMM helper: 256x2048 @ 2048x128, both m-tiles -------
// B (global) software-pipelined 3 tiles ahead.
__device__ __forceinline__ void gemm2048_2mt(
    const uint16_t* __restrict__ Wpk, const uint16_t* const* sbase,
    int wave, int lhalf, int l16, v8f acc0[8], v8f acc1[8])
{
#pragma unroll
  for (int i = 0; i < 8; ++i) { acc0[i] = zero8(); acc1[i] = zero8(); }
  const int arow0 = (wave * 2) * 16 + l16;
  const uint16_t* wr0 = Wpk + (size_t)arow0 * 2048 + lhalf * 16;
  const uint16_t* wr1 = wr0 + (size_t)16 * 2048;
#pragma unroll 4
  for (int kb = 0; kb < 2048; kb += 32) {
    __builtin_prefetch(wr0 + kb + 128, 0, 1);
    __builtin_prefetch(wr1 + kb + 128, 0, 1);
    v16bf A0 = loadAB(wr0 + kb, wr0 + kb + 8);
    v16bf A1 = loadAB(wr1 + kb, wr1 + kb + 8);
    v16bf B0 = loadAB(sbase[0] + kb, sbase[0] + kb + 8);
    v16bf B1 = loadAB(sbase[1] + kb, sbase[1] + kb + 8);
    v16bf B2 = loadAB(sbase[2] + kb, sbase[2] + kb + 8);
#pragma unroll
    for (int nt = 0; nt < 8; ++nt) {
      v16bf Bn = B2;
      if (nt < 5) Bn = loadAB(sbase[nt + 3] + kb, sbase[nt + 3] + kb + 8);
      acc0[nt] = wmma_bf16(A0, B0, acc0[nt]);
      acc1[nt] = wmma_bf16(A1, B0, acc1[nt]);
      B0 = B1;
      B1 = B2;
      B2 = Bn;
    }
  }
}

// ---------------- kernel 3: fusion + preds + final output ----------------
__global__ __launch_bounds__(256) void snake_net_tail(
    const uint16_t* __restrict__ stateT,  // [64][128][2048] bf16
    const uint16_t* __restrict__ WfPk,    // [256][2048] bf16
    const float*    __restrict__ fusb,    // [256]
    const uint16_t* __restrict__ W0sPk,   // [256][2048] bf16 (state part)
    const float*    __restrict__ w0full,  // [256][2304] f32 (for g part)
    const float*    __restrict__ b0,      // [256]
    const uint16_t* __restrict__ W1Pk,    // [64][256] bf16
    const float*    __restrict__ b1,      // [64]
    const float*    __restrict__ w2,      // [2][64] f32
    const float*    __restrict__ b2,      // [2]
    const float*    __restrict__ snakes,  // [64][128][2]
    float*          __restrict__ out)     // [64][128][2]
{
  extern __shared__ unsigned char smemraw[];
  float*    g   = (float*)smemraw;                 // 256
  float*    h0c = g + 256;                         // 256
  uint16_t* h1T = (uint16_t*)(h0c + 256);          // [128][256]
  uint16_t* h2T = h1T + 128 * 256;                 // [128][64]

  const int t = blockIdx.x, tid = threadIdx.x;
  const int wave = tid >> 5, lane = tid & 31;
  const int lhalf = lane >> 4, l16 = lane & 15;
  const uint16_t* S = stateT + (size_t)t * 128 * 2048;

  const uint16_t* sbase[8];
#pragma unroll
  for (int nt = 0; nt < 8; ++nt)
    sbase[nt] = S + (size_t)(nt * 16 + l16) * 2048 + lhalf * 16;

  g[tid] = -3.4e38f;
  __syncthreads();

  // ---- fusion: 256x2048 @ 2048x128, then max over positions -> g[256] ----
  {
    v8f acc0[8], acc1[8];
    gemm2048_2mt(WfPk, sbase, wave, lhalf, l16, acc0, acc1);
#pragma unroll
    for (int mi = 0; mi < 2; ++mi) {
      const v8f* acc = mi ? acc1 : acc0;
      const int mrow = (wave * 2 + mi) * 16 + lhalf * 8;
      float mx[8];
#pragma unroll
      for (int r = 0; r < 8; ++r) mx[r] = -3.4e38f;
#pragma unroll
      for (int nt = 0; nt < 8; ++nt)
#pragma unroll
        for (int r = 0; r < 8; ++r)
          mx[r] = fmaxf(mx[r], acc[nt][r] + fusb[mrow + r]);
#pragma unroll
      for (int off = 1; off < 16; off <<= 1)
#pragma unroll
        for (int r = 0; r < 8; ++r) mx[r] = fmaxf(mx[r], __shfl_xor(mx[r], off));
      if (l16 == 0)
#pragma unroll
        for (int r = 0; r < 8; ++r) atomicMaxF(&g[mrow + r], mx[r]);
    }
  }
  __syncthreads();

  // ---- pred0 column-constant part: h0c[o] = W0[o, 2048..2303] . g + b0 ----
  {
    const int o = tid;
    const float* wr = w0full + (size_t)o * 2304 + 2048;
    float s = b0[o];
    for (int c = 0; c < 256; ++c) s += wr[c] * g[c];
    h0c[o] = s;
  }
  __syncthreads();

  // ---- pred0 main GEMM: 256x2048 @ 2048x128, +h0c, relu -> h1T ----
  {
    v8f acc0[8], acc1[8];
    gemm2048_2mt(W0sPk, sbase, wave, lhalf, l16, acc0, acc1);
#pragma unroll
    for (int mi = 0; mi < 2; ++mi) {
      const v8f* acc = mi ? acc1 : acc0;
      const int mrow = (wave * 2 + mi) * 16 + lhalf * 8;
#pragma unroll
      for (int nt = 0; nt < 8; ++nt) {
        const int p = nt * 16 + l16;
        float v[8];
#pragma unroll
        for (int r = 0; r < 8; ++r)
          v[r] = fmaxf(acc[nt][r] + h0c[mrow + r], 0.0f);
        *(uint4*)&h1T[p * 256 + mrow] = pack8(v);
      }
    }
  }
  __syncthreads();

  // ---- pred1: 64x256 @ 256x128, relu -> h2T ----
  {
    const int mt  = wave & 3;
    const int ntb = (wave >> 2) * 4;
    v8f acc[4];
#pragma unroll
    for (int i = 0; i < 4; ++i) acc[i] = zero8();
    const int arow = mt * 16 + l16;
    const uint16_t* wr = W1Pk + (size_t)arow * 256 + lhalf * 16;
    int hoff4[4];
#pragma unroll
    for (int q = 0; q < 4; ++q)
      hoff4[q] = ((ntb + q) * 16 + l16) * 256 + lhalf * 16;
#pragma unroll
    for (int kb = 0; kb < 256; kb += 32) {
      v16bf A = loadAB(wr + kb, wr + kb + 8);
#pragma unroll
      for (int q = 0; q < 4; ++q) {
        v16bf B = loadAB(&h1T[hoff4[q] + kb], &h1T[hoff4[q] + kb + 8]);
        acc[q] = wmma_bf16(A, B, acc[q]);
      }
    }
    const int mrow = mt * 16 + lhalf * 8;
#pragma unroll
    for (int q = 0; q < 4; ++q) {
      const int p = (ntb + q) * 16 + l16;
      float v[8];
#pragma unroll
      for (int r = 0; r < 8; ++r) v[r] = fmaxf(acc[q][r] + b1[mrow + r], 0.0f);
      *(uint4*)&h2T[p * 64 + mrow] = pack8(v);
    }
  }
  __syncthreads();

  // ---- pred2 (2x64) + final: out = snakes + offset ----
  {
    const int p = tid >> 1, j = tid & 1;
    const float* wr = w2 + j * 64;
    float s = b2[j];
    for (int c = 0; c < 64; ++c) s += wr[c] * bf2f(h2T[p * 64 + c]);
    const size_t o = (size_t)t * 256 + p * 2 + j;
    out[o] = snakes[o] + s;
  }
}

// ---------------- host launcher ----------------

extern "C" void kernel_launch(void* const* d_in, const int* in_sizes, int n_in,
                              void* d_out, int out_size, void* d_ws, size_t ws_size,
                              hipStream_t stream) {
  (void)in_sizes; (void)n_in; (void)out_size; (void)ws_size;

  const float* cnn    = (const float*)d_in[0];
  const float* snakes = (const float*)d_in[1];
  const float* fus_b  = (const float*)d_in[2];
  const float* fus_w  = (const float*)d_in[3];
  const float* h_b    = (const float*)d_in[4];
  const float* h_beta = (const float*)d_in[5];
  const float* h_gam  = (const float*)d_in[6];
  const float* h_mean = (const float*)d_in[7];
  const float* h_var  = (const float*)d_in[8];
  const float* h_w    = (const float*)d_in[9];
  const float* p0_b   = (const float*)d_in[10];
  const float* p0_w   = (const float*)d_in[11];
  const float* p1_b   = (const float*)d_in[12];
  const float* p1_w   = (const float*)d_in[13];
  const float* p2_b   = (const float*)d_in[14];
  const float* p2_w   = (const float*)d_in[15];
  const float *r_b[7], *r_beta[7], *r_gam[7], *r_mean[7], *r_var[7], *r_w[7];
  for (int i = 0; i < 7; ++i) {
    const int base = 16 + 6 * i;
    r_b[i]    = (const float*)d_in[base + 0];
    r_beta[i] = (const float*)d_in[base + 1];
    r_gam[i]  = (const float*)d_in[base + 2];
    r_mean[i] = (const float*)d_in[base + 3];
    r_var[i]  = (const float*)d_in[base + 4];
    r_w[i]    = (const float*)d_in[base + 5];
  }

  // workspace carve-up (256B aligned)
  uint8_t* ws = (uint8_t*)d_ws;
  size_t off = 0;
  auto take = [&](size_t bytes) -> void* {
    void* p = ws + off;
    off = (off + bytes + 255) & ~(size_t)255;
    return p;
  };
  uint16_t* initT  = (uint16_t*)take((size_t)64 * 128 * 256 * 2);   // 4 MB
  uint16_t* stateT = (uint16_t*)take((size_t)64 * 128 * 2048 * 2);  // 32 MB
  uint16_t* WhPk   = (uint16_t*)take((size_t)256 * 1440 * 2);
  uint16_t* WrPk   = (uint16_t*)take((size_t)7 * 256 * 2304 * 2);
  uint16_t* WfPk   = (uint16_t*)take((size_t)256 * 2048 * 2);
  uint16_t* W0sPk  = (uint16_t*)take((size_t)256 * 2048 * 2);
  uint16_t* W1Pk   = (uint16_t*)take((size_t)64 * 256 * 2);
  float*    bnp    = (float*)take((size_t)8 * 3 * 256 * 4);

  // ---- prep ----
  pack_bn<<<1, 256, 0, stream>>>(h_b, h_gam, h_beta, h_mean, h_var, bnp);
  for (int i = 0; i < 7; ++i)
    pack_bn<<<1, 256, 0, stream>>>(r_b[i], r_gam[i], r_beta[i], r_mean[i], r_var[i],
                                   bnp + (size_t)(i + 1) * 768);
  pack_conv9<<<(256 * 9 * 160 + 255) / 256, 256, 0, stream>>>(h_w, WhPk, 130, 160);
  for (int i = 0; i < 7; ++i)
    pack_conv9<<<(256 * 9 * 256 + 255) / 256, 256, 0, stream>>>(
        r_w[i], WrPk + (size_t)i * 256 * 2304, 256, 256);
  pack_1x1<<<(256 * 2048 + 255) / 256, 256, 0, stream>>>(fus_w, WfPk, 2048, 2048, 256 * 2048);
  pack_1x1<<<(256 * 2048 + 255) / 256, 256, 0, stream>>>(p0_w, W0sPk, 2048, 2304, 256 * 2048);
  pack_1x1<<<(64 * 256 + 255) / 256, 256, 0, stream>>>(p1_w, W1Pk, 256, 256, 64 * 256);

  // ---- main pipeline ----
  sample_kernel<<<64, 256, 0, stream>>>(cnn, snakes, initT);

  const size_t lds2 = (size_t)2 * HROWS * 256 * 2;                     // 160 KB
  snake_net_blocks<<<64, 256, lds2, stream>>>(initT, WhPk, WrPk, bnp, stateT);

  const size_t lds3 = (size_t)(256 + 256) * 4 + (size_t)128 * 256 * 2 +
                      (size_t)128 * 64 * 2;                            // ~82 KB
  snake_net_tail<<<64, 256, lds3, stream>>>(stateT, WfPk, fus_b, W0sPk, p0_w, p0_b,
                                            W1Pk, p1_b, p2_w, p2_b, snakes,
                                            (float*)d_out);
}
